// HierarchicalDenseCRF_50714973831454
// MI455X (gfx1250) — matrix operand order repaired and checked by reading, not compiled, not run on hardware
//
#include <hip/hip_runtime.h>
#include <hip/hip_bf16.h>
#include <math.h>

typedef _Float16 v16h __attribute__((ext_vector_type(16)));
typedef float    v8f  __attribute__((ext_vector_type(8)));

#define NCLS 8
#define GCH 6          // guidance channels
#define INCH 14        // NCLS + GCH
#define KPAD 128       // 126 real taps padded to 128
#define MF_TS 16       // tile side (pixels)
#define MF_LW 18       // tile + halo
#define MF_PL (MF_LW*MF_LW)

// ---------------------------------------------------------------------------
// Fold spatial (depthwise 3x3) + bilateral (14->8 3x3) + compat (8x8 1x1)
// into one 8x14x3x3 conv, stored padded as Wpad[8][128].
// ---------------------------------------------------------------------------
__global__ void fold_weights_kernel(const float* __restrict__ spatial,
                                    const float* __restrict__ bilat,
                                    const float* __restrict__ compat,
                                    float* __restrict__ Wpad) {
  int idx = blockIdx.x * 256 + threadIdx.x;
  if (idx >= NCLS * KPAD) return;
  int o = idx >> 7;
  int k = idx & (KPAD - 1);
  float v = 0.f;
  if (k < INCH * 9) {
    int c  = k / 9;
    int r9 = k - c * 9;
    #pragma unroll
    for (int i = 0; i < NCLS; ++i)
      v += compat[o * NCLS + i] * bilat[(i * INCH + c) * 9 + r9];
    if (c < NCLS)
      v += compat[o * NCLS + c] * spatial[c * 9 + r9];
  }
  Wpad[idx] = v;
}

// ---------------------------------------------------------------------------
// 1x1 projection of a skip tensor to 3 channels (+bias). One thread per pixel
// computes all 3 outputs -> skip tensor is read exactly once.
// ---------------------------------------------------------------------------
__global__ void proj_kernel(const float* __restrict__ skip,
                            const float* __restrict__ w,
                            const float* __restrict__ bias,
                            float* __restrict__ out,
                            int CH, int HW, int N) {
  int idx = blockIdx.x * 256 + threadIdx.x;
  if (idx >= N) return;
  int b = idx / HW;
  int p = idx - b * HW;
  const float* sp = skip + (size_t)b * CH * HW + p;
  float a0 = 0.f, a1 = 0.f, a2 = 0.f;
  for (int c = 0; c < CH; ++c) {
    float v = sp[(size_t)c * HW];
    a0 += v * w[c];
    a1 += v * w[CH + c];
    a2 += v * w[2 * CH + c];
  }
  float* dst = out + (size_t)b * 3 * HW + p;
  dst[0]            = a0 + bias[0];
  dst[(size_t)HW]   = a1 + bias[1];
  dst[(size_t)2*HW] = a2 + bias[2];
}

// align_corners bilinear sample on one channel plane
__device__ __forceinline__ float bilerp_plane(const float* __restrict__ p,
                                              int IH, int IW, float yc, float xc) {
  float y0f = floorf(yc), x0f = floorf(xc);
  int iy0 = (int)y0f, ix0 = (int)x0f;
  iy0 = iy0 < 0 ? 0 : (iy0 > IH - 1 ? IH - 1 : iy0);
  ix0 = ix0 < 0 ? 0 : (ix0 > IW - 1 ? IW - 1 : ix0);
  int iy1 = iy0 + 1 < IH ? iy0 + 1 : IH - 1;
  int ix1 = ix0 + 1 < IW ? ix0 + 1 : IW - 1;
  float wy = yc - y0f, wx = xc - x0f;
  float a = p[(size_t)iy0 * IW + ix0], b = p[(size_t)iy0 * IW + ix1];
  float c = p[(size_t)iy1 * IW + ix0], d = p[(size_t)iy1 * IW + ix1];
  float top = a * (1.f - wx) + b * wx;
  float bot = c * (1.f - wx) + d * wx;
  return top * (1.f - wy) + bot * wy;
}

// ---------------------------------------------------------------------------
// guidance[b][0..2] = image; guidance[b][3..5] = mean of 4 resized projections
// ---------------------------------------------------------------------------
__global__ void guidance_kernel(const float* __restrict__ image,
                                const float* __restrict__ p0,
                                const float* __restrict__ p1,
                                const float* __restrict__ p2,
                                const float* __restrict__ p3,
                                float* __restrict__ guid,
                                int H, int W, int N) {
  int idx = blockIdx.x * 256 + threadIdx.x;
  if (idx >= N) return;
  int HW = H * W;
  int b = idx / HW;
  int p = idx - b * HW;
  int y = p / W, x = p - (p / W) * W;
  const float sc0 = 255.f / 255.f, sc1 = 127.f / 255.f,
              sc2 = 63.f / 255.f,  sc3 = 31.f / 255.f;
  #pragma unroll
  for (int j = 0; j < 3; ++j) {
    guid[((size_t)(b * GCH + j)) * HW + p] = image[((size_t)(b * 3 + j)) * HW + p];
    float s = 0.f;
    s += bilerp_plane(p0 + (size_t)(b * 3 + j) * 65536, 256, 256, y * sc0, x * sc0);
    s += bilerp_plane(p1 + (size_t)(b * 3 + j) * 16384, 128, 128, y * sc1, x * sc1);
    s += bilerp_plane(p2 + (size_t)(b * 3 + j) * 4096,   64,  64, y * sc2, x * sc2);
    s += bilerp_plane(p3 + (size_t)(b * 3 + j) * 1024,   32,  32, y * sc3, x * sc3);
    guid[((size_t)(b * GCH + 3 + j)) * HW + p] = s * 0.25f;
  }
}

// generic align_corners bilinear resize, NCHW, one thread per output element
__global__ void resize_kernel(const float* __restrict__ in, float* __restrict__ out,
                              int IH, int IW, int OH, int OW, int N) {
  int idx = blockIdx.x * 256 + threadIdx.x;
  if (idx >= N) return;
  int OHW = OH * OW;
  int plane = idx / OHW;
  int p = idx - plane * OHW;
  int oy = p / OW, ox = p - (p / OW) * OW;
  float sy = (OH > 1) ? (float)(IH - 1) / (float)(OH - 1) : 0.f;
  float sx = (OW > 1) ? (float)(IW - 1) / (float)(OW - 1) : 0.f;
  out[idx] = bilerp_plane(in + (size_t)plane * IH * IW, IH, IW, oy * sy, ox * sx);
}

// channel softmax over 8 classes, NCHW
__global__ void softmax8_kernel(const float* __restrict__ in, float* __restrict__ out,
                                int HW, int N) {
  int idx = blockIdx.x * 256 + threadIdx.x;
  if (idx >= N) return;
  int b = idx / HW;
  int p = idx - b * HW;
  const float* src = in + (size_t)b * NCLS * HW + p;
  float v[NCLS], mx = -3.4e38f;
  #pragma unroll
  for (int c = 0; c < NCLS; ++c) { v[c] = src[(size_t)c * HW]; mx = fmaxf(mx, v[c]); }
  float s = 0.f;
  #pragma unroll
  for (int c = 0; c < NCLS; ++c) { v[c] = expf(v[c] - mx); s += v[c]; }
  float inv = 1.f / s;
  float* dst = out + (size_t)b * NCLS * HW + p;
  #pragma unroll
  for (int c = 0; c < NCLS; ++c) dst[(size_t)c * HW] = v[c] * inv;
}

// ---------------------------------------------------------------------------
// Fused mean-field iteration via WMMA implicit GEMM.
//   message = conv3x3(concat(Q,guidance), W_total)   (K=126 pad 128, N=8 pad 16)
//   Qout    = softmax(logits - message)
// 256 threads = 8 waves per 16x16 pixel tile; wave w computes rows 2w, 2w+1.
// ---------------------------------------------------------------------------
__global__ __launch_bounds__(256)
void meanfield_kernel(const float* __restrict__ Qin,
                      const float* __restrict__ guid,
                      const float* __restrict__ logits,
                      const float* __restrict__ Wpad,
                      float* __restrict__ Qout,
                      int H, int W) {
  __shared__ float smP[INCH * MF_PL];   // 14 channel planes, 18x18 each
  __shared__ int   smOff[KPAD];         // tap k -> plane offset (c*324+ky*18+kx)

  const int tid = threadIdx.x;
  const int b   = blockIdx.z;
  const int x0  = blockIdx.x * MF_TS;
  const int y0  = blockIdx.y * MF_TS;
  const int HW  = H * W;

  // stage tile + halo (zero pad) from Q (ch 0..7) and guidance (ch 8..13)
  for (int i = tid; i < INCH * MF_PL; i += 256) {
    int c   = i / MF_PL;
    int rem = i - c * MF_PL;
    int ty  = rem / MF_LW;
    int tx  = rem - ty * MF_LW;
    int gy  = y0 - 1 + ty;
    int gx  = x0 - 1 + tx;
    float v = 0.f;
    if (gy >= 0 && gy < H && gx >= 0 && gx < W) {
      v = (c < NCLS) ? Qin [((size_t)(b * NCLS + c))       * HW + (size_t)gy * W + gx]
                     : guid[((size_t)(b * GCH + (c - 8)))  * HW + (size_t)gy * W + gx];
    }
    smP[i] = v;
  }
  if (tid < KPAD) {
    int k = tid, off = 0;
    if (k < INCH * 9) {
      int c = k / 9, r9 = k - c * 9, ky = r9 / 3, kx = r9 - ky * 3;
      off = c * MF_PL + ky * MF_LW + kx;
    }
    smOff[tid] = off;   // k>=126 -> offset 0 (value killed by zero B weight)
  }
  __syncthreads();

  const int lane = tid & 31;
  const int wave = tid >> 5;
  const int hi   = lane >> 4;      // half-wave
  const int m    = lane & 15;      // A: pixel-x in strip | B/C: class column N

  // B fragments from folded weights: lanes 0-15 hold K {0..7,16..23}; 16-31 hold {8..15,24..31}
  v16h bf[4];
  #pragma unroll
  for (int s = 0; s < 4; ++s) {
    #pragma unroll
    for (int e = 0; e < 16; ++e) {
      int kg = 32 * s + e + ((e < 8) ? 0 : 8) + (hi ? 8 : 0);
      float wv = (m < NCLS) ? Wpad[m * KPAD + kg] : 0.f;
      bf[s][e] = (_Float16)wv;
    }
  }

  #pragma unroll
  for (int rr = 0; rr < 2; ++rr) {
    const int yl    = 2 * wave + rr;          // tile row 0..15
    const int ybase = yl * MF_LW + m;         // LDS base for this lane's pixel
    v8f acc = {0.f, 0.f, 0.f, 0.f, 0.f, 0.f, 0.f, 0.f};
    #pragma unroll
    for (int s = 0; s < 4; ++s) {
      // A 16x32 f16 layout: lanes 0-15 -> K {0..7,16..23}; 16-31 -> {8..15,24..31}
      v16h a;
      #pragma unroll
      for (int e = 0; e < 16; ++e) {
        int koff = e + ((e < 8) ? 0 : 8) + (hi ? 8 : 0);
        int kg   = 32 * s + koff;
        a[e] = (_Float16)smP[smOff[kg] + ybase];
      }
      acc = __builtin_amdgcn_wmma_f32_16x16x32_f16(
          false, a, false, bf[s], (short)0, acc, false, false);
    }
    // D layout: lane (m + hi*16) holds class n=m, pixels M = r + hi*8 in acc[r]
    const int y = y0 + yl;
    #pragma unroll
    for (int r = 0; r < 8; ++r) {
      const int px  = x0 + r + (hi ? 8 : 0);
      float lg  = (m < NCLS)
                ? logits[((size_t)(b * NCLS + m)) * HW + (size_t)y * W + px] : 0.f;
      float val = lg - acc[r];
      float mx = val;
      mx = fmaxf(mx, __shfl_xor(mx, 1, 32));
      mx = fmaxf(mx, __shfl_xor(mx, 2, 32));
      mx = fmaxf(mx, __shfl_xor(mx, 4, 32));
      float e1 = expf(val - mx);
      float sm = e1;
      sm += __shfl_xor(sm, 1, 32);
      sm += __shfl_xor(sm, 2, 32);
      sm += __shfl_xor(sm, 4, 32);
      float q = e1 / sm;
      if (m < NCLS)
        Qout[((size_t)(b * NCLS + m)) * HW + (size_t)y * W + px] = q;
    }
  }
}

// out = 0.5 * (Qfull + upsample(Qdown))
__global__ void combine_kernel(const float* __restrict__ Qf,
                               const float* __restrict__ Qd,
                               float* __restrict__ out,
                               int H, int W, int DH, int DW, int N) {
  int idx = blockIdx.x * 256 + threadIdx.x;
  if (idx >= N) return;
  int HW = H * W;
  int plane = idx / HW;
  int p = idx - plane * HW;
  int y = p / W, x = p - (p / W) * W;
  float sy = (float)(DH - 1) / (float)(H - 1);
  float sx = (float)(DW - 1) / (float)(W - 1);
  float up = bilerp_plane(Qd + (size_t)plane * DH * DW, DH, DW, y * sy, x * sx);
  out[idx] = 0.5f * (Qf[idx] + up);
}

// ---------------------------------------------------------------------------
extern "C" void kernel_launch(void* const* d_in, const int* in_sizes, int n_in,
                              void* d_out, int out_size, void* d_ws, size_t ws_size,
                              hipStream_t stream) {
  (void)in_sizes; (void)n_in; (void)out_size; (void)ws_size;
  const float* logits  = (const float*)d_in[0];
  const float* image   = (const float*)d_in[1];
  const float* skip0   = (const float*)d_in[2];
  const float* skip1   = (const float*)d_in[3];
  const float* skip2   = (const float*)d_in[4];
  const float* skip3   = (const float*)d_in[5];
  const float* spat_w  = (const float*)d_in[6];
  const float* bilat_w = (const float*)d_in[7];
  const float* compat  = (const float*)d_in[8];
  const float* sp_w[4] = {(const float*)d_in[9],  (const float*)d_in[11],
                          (const float*)d_in[13], (const float*)d_in[15]};
  const float* sp_b[4] = {(const float*)d_in[10], (const float*)d_in[12],
                          (const float*)d_in[14], (const float*)d_in[16]};
  float* out = (float*)d_out;

  const int B = 8, H = 256, W = 256, HW = H * W;
  const int DH = 128, DW = 128, DHW = DH * DW;

  // workspace carve-up (floats)
  float* ws = (float*)d_ws;
  size_t o = 0;
  float* proj0 = ws + o; o += (size_t)B * 3 * 65536;
  float* proj1 = ws + o; o += (size_t)B * 3 * 16384;
  float* proj2 = ws + o; o += (size_t)B * 3 * 4096;
  float* proj3 = ws + o; o += (size_t)B * 3 * 1024;
  float* guid  = ws + o; o += (size_t)B * GCH * HW;
  float* guidd = ws + o; o += (size_t)B * GCH * DHW;
  float* logd  = ws + o; o += (size_t)B * NCLS * DHW;
  float* qa    = ws + o; o += (size_t)B * NCLS * HW;
  float* qb    = ws + o; o += (size_t)B * NCLS * HW;
  float* qad   = ws + o; o += (size_t)B * NCLS * DHW;
  float* qbd   = ws + o; o += (size_t)B * NCLS * DHW;
  float* wpad  = ws + o; o += (size_t)NCLS * KPAD;

  // 1) fold spatial+bilateral+compat into one padded weight matrix
  fold_weights_kernel<<<4, 256, 0, stream>>>(spat_w, bilat_w, compat, wpad);

  // 2) skip projections (each skip tensor read once)
  { int N0 = B * 65536; proj_kernel<<<(N0+255)/256,256,0,stream>>>(skip0, sp_w[0], sp_b[0], proj0,  64, 65536, N0); }
  { int N1 = B * 16384; proj_kernel<<<(N1+255)/256,256,0,stream>>>(skip1, sp_w[1], sp_b[1], proj1, 128, 16384, N1); }
  { int N2 = B * 4096;  proj_kernel<<<(N2+255)/256,256,0,stream>>>(skip2, sp_w[2], sp_b[2], proj2, 256,  4096, N2); }
  { int N3 = B * 1024;  proj_kernel<<<(N3+255)/256,256,0,stream>>>(skip3, sp_w[3], sp_b[3], proj3, 512,  1024, N3); }

  // 3) guidance = concat(image, mean of resized projections)
  { int N = B * HW;
    guidance_kernel<<<(N+255)/256,256,0,stream>>>(image, proj0, proj1, proj2, proj3, guid, H, W, N); }

  // 4) downsampled logits & guidance (align_corners 256->128)
  { int N = B * NCLS * DHW; resize_kernel<<<(N+255)/256,256,0,stream>>>(logits, logd, H, W, DH, DW, N); }
  { int N = B * GCH  * DHW; resize_kernel<<<(N+255)/256,256,0,stream>>>(guid,  guidd, H, W, DH, DW, N); }

  // 5) initial softmax
  { int N = B * HW;  softmax8_kernel<<<(N+255)/256,256,0,stream>>>(logits, qa,  HW,  N); }
  { int N = B * DHW; softmax8_kernel<<<(N+255)/256,256,0,stream>>>(logd,   qad, DHW, N); }

  // 6) mean-field iterations (WMMA fused message+softmax)
  dim3 gFull(W / MF_TS, H / MF_TS, B);
  meanfield_kernel<<<gFull, 256, 0, stream>>>(qa, guid, logits, wpad, qb, H, W);
  meanfield_kernel<<<gFull, 256, 0, stream>>>(qb, guid, logits, wpad, qa, H, W);

  dim3 gDown(DW / MF_TS, DH / MF_TS, B);
  meanfield_kernel<<<gDown, 256, 0, stream>>>(qad, guidd, logd, wpad, qbd, DH, DW);
  meanfield_kernel<<<gDown, 256, 0, stream>>>(qbd, guidd, logd, wpad, qad, DH, DW);

  // 7) combine full-res and upsampled half-res results
  { int N = B * NCLS * HW;
    combine_kernel<<<(N+255)/256,256,0,stream>>>(qa, qad, out, H, W, DH, DW, N); }
}
